// ControlledConvEMAStabilizer_52664888983904
// MI455X (gfx1250) — matrix operand order repaired and checked by reading, not compiled, not run on hardware
//
#include <hip/hip_runtime.h>
#include <hip/hip_bf16.h>
#include <stdint.h>

// ---------------------------------------------------------------------------
// ControlledConvEMAStabilizer for MI455X (gfx1250, wave32).
// Conv layers = implicit GEMM on v_wmma_f32_16x16x32_f16:
//   wave tile: M = 16 spatial (along W) x N = 16*NT Cout (NT accumulators),
//   K = Cin*9 in 32-wide chunks.
// A: async global->LDS halo rows (clamped addrs, masked at fragment build).
// B: Tensor Data Mover 3-D tile (1 x 32 Cin x 16*NT Cout, stride-9 gather)
//    into LDS [n][kc] layout -> contiguous ds_load_b128 fragment reads.
// All NT B-fragments are materialized before the WMMA burst to avoid
// WMMA->VALU WAR hazard nops and ds_load serialization.
// ---------------------------------------------------------------------------

typedef __attribute__((ext_vector_type(16))) _Float16 v16h;
typedef __attribute__((ext_vector_type(8)))  float    v8f;
typedef __attribute__((ext_vector_type(4)))  unsigned uint32x4;
typedef __attribute__((ext_vector_type(8)))  int      int32x8;
typedef __attribute__((ext_vector_type(4)))  int      int32x4;

#define WAVES 4
#define HDIM  128
#define WDIM  128
#define HWDIM (HDIM * WDIM)

// Tensor Data Mover: toolchain-dependent builtin arity (see CDNA5_HIP.md).
#if __has_include(<hip/amd_detail/amd_gfx1250_TDM.h>)
#define TDM_LOAD(g0, g1, g2, g3) \
    __builtin_amdgcn_tensor_load_to_lds((g0), (g1), (g2), (g3), (int32x8)0, 0)
#else
#define TDM_LOAD(g0, g1, g2, g3) \
    __builtin_amdgcn_tensor_load_to_lds((g0), (g1), (g2), (g3), 0)
#endif

// conv0 reads [backbone(64) | z(32) | mem_stab(32) | mem_unstab(32)] without
// materializing the concat; other layers are single-segment (branch-free).
__device__ __forceinline__ const float* chan_base(
    const float* s0, const float* s1, const float* s2, const float* s3,
    int c0n, int c1n, int c2n, int b, int c)
{
    const float* s;
    int cc = c, segC;
    if (cc < c0n)               { s = s0; segC = c0n; }
    else if ((cc -= c0n) < c1n) { s = s1; segC = c1n; }
    else if ((cc -= c1n) < c2n) { s = s2; segC = c2n; }
    else                        { cc -= c2n; s = s3; segC = 32; }
    return s + ((size_t)b * segC + cc) * (size_t)HWDIM;
}

template <bool MULTI, int NT>
__global__ __launch_bounds__(WAVES * 32)
void conv3x3_wmma(const float* __restrict__ s0, const float* __restrict__ s1,
                  const float* __restrict__ s2, const float* __restrict__ s3,
                  int c0n, int c1n, int c2n,
                  const float* __restrict__ wgt,   // [Cout, Cin, 3, 3]
                  const float* __restrict__ bias,  // [Cout]
                  float* __restrict__ out,         // [B, Cout, H, W]
                  int Cin, int Cout, int leaky)
{
    __shared__ float lA[WAVES][32][18];       // [kc][halo m], f32, per-wave
    __shared__ float lB[WAVES][16 * NT][32];  // [n][kc] (TDM write order), f32

    const int tid  = threadIdx.x;
    const int wv   = tid >> 5;
    const int lane = tid & 31;
    const int l    = lane & 15;
    const int hi   = lane >> 4;

    const int b  = blockIdx.z;
    const int n0 = blockIdx.y * (16 * NT);           // NT Cout tiles per wave
    const int m0 = (blockIdx.x * WAVES + wv) * 16;   // spatial tile base
    const int h  = m0 >> 7;
    const int w0 = m0 & (WDIM - 1);

    const unsigned ldsA = (unsigned)(uintptr_t)&lA[wv][lane][0];
    const unsigned ldsB = (unsigned)(uintptr_t)&lB[wv][0][0];

    // Loop-invariant TDM descriptor parts (only global_addr varies).
    int32x8 g1;
    g1[0] = 0x00020000;               // data_size=4B
    g1[1] = 0x00010000;               // tensor_dim0=1
    g1[2] = 0x00200000;               // tensor_dim1=32
    g1[3] = 0x00010000;               // tile_dim0=1
    g1[4] = ((16 * NT) << 16) | 32;   // tile_dim2=16*NT, tile_dim1=32
    g1[5] = 9;                        // tensor_dim0_stride=9 elements
    g1[6] = (Cin * 9) << 16;          // tensor_dim1_stride lo
    g1[7] = 0;                        // tensor_dim1_stride hi
    int32x4 g2; g2[0] = 16 * NT; g2[1] = 0; g2[2] = 0; g2[3] = 0;
    int32x4 g3; g3[0] = 0; g3[1] = 0; g3[2] = 0; g3[3] = 0;

    v8f acc[NT];
#pragma unroll
    for (int t = 0; t < NT; ++t) acc[t] = (v8f){};

    for (int kh = 0; kh < 3; ++kh) {
        const int  hp  = h + kh - 1;
        const bool hok = (unsigned)hp < (unsigned)HDIM;
        const int  hc  = hp < 0 ? 0 : (hp > HDIM - 1 ? HDIM - 1 : hp);

        for (int c0 = 0; c0 < Cin; c0 += 32) {
            // ---- stage A: 18-wide halo row per channel via async DMA ------
            const float* base =
                MULTI ? chan_base(s0, s1, s2, s3, c0n, c1n, c2n, b, c0 + lane)
                      : s0 + ((size_t)b * Cin + (c0 + lane)) * (size_t)HWDIM;
            const float* row = base + (size_t)hc * WDIM;

            // prior iteration's ds_loads of lA must drain before overwrite
            asm volatile("s_wait_dscnt 0x0" ::: "memory");
#pragma unroll
            for (int m = 0; m < 18; ++m) {
                const int wp = w0 - 1 + m;
                const int wc = wp < 0 ? 0 : (wp > WDIM - 1 ? WDIM - 1 : wp);
                const float* gp = row + wc;          // clamped: always valid
                const unsigned la = ldsA + (unsigned)(m * 4);
                asm volatile("global_load_async_to_lds_b32 %0, %1, off"
                             :: "v"(la), "v"(gp) : "memory");
            }
            if (c0 + 32 < Cin)  // speculative prefetch of next K-chunk row
                __builtin_prefetch(row + (size_t)32 * HWDIM, 0, 1);
            asm volatile("s_wait_asynccnt 0x0" ::: "memory");

            for (int kw = 0; kw < 3; ++kw) {
                const int tap = kh * 3 + kw;

                // ---- stage B: one TDM tile, 1 x 32(kc) x 16*NT(n) ---------
                {
                    const uint64_t ga = (uint64_t)(uintptr_t)
                        (wgt + ((size_t)n0 * Cin + c0) * 9 + tap);
                    uint32x4 g0;
                    g0[0] = 1u;                       // count=1 valid D#
                    g0[1] = ldsB;                     // lds_addr
                    g0[2] = (unsigned)ga;             // global_addr[31:0]
                    g0[3] = (unsigned)(ga >> 32) | 0x80000000u; // [56:32]|type=2

                    // prior ds_loads of lB must drain before TDM rewrites it
                    asm volatile("s_wait_dscnt 0x0" ::: "memory");
                    TDM_LOAD(g0, g1, g2, g3);
                    __builtin_amdgcn_s_wait_tensorcnt(0);
                }

                // ---- A fragment (built once, reused by NT WMMAs) ----------
                // lane l = spatial row M=l; halo column = l+kw; validity
                // depends only on (hp, wp) -> cndmask, no branch.
                const int  wp2 = w0 + kw - 1 + l;
                const bool vm  = hok && (unsigned)wp2 < (unsigned)WDIM;
                const float* ap = &lA[wv][0][l + kw];
                v16h af;
#pragma unroll
                for (int j = 0; j < 8; ++j) {
                    const float a0 = ap[(hi * 8 + j) * 18];       // K=hi*8+j
                    const float a1 = ap[(16 + hi * 8 + j) * 18];  // K=16+hi*8+j
                    af[j]     = vm ? (_Float16)a0 : (_Float16)0.0f;
                    af[j + 8] = vm ? (_Float16)a1 : (_Float16)0.0f;
                }

                // ---- build ALL NT B fragments first (distinct registers,
                //      overlapping ds_load_b128s), then WMMA burst ----------
                v16h bf[NT];
#pragma unroll
                for (int t = 0; t < NT; ++t) {
                    const float4* bp =
                        (const float4*)&lB[wv][t * 16 + l][hi * 8];
                    const float4 x0 = bp[0];
                    const float4 x1 = bp[1];
                    const float4 x2 = bp[4];
                    const float4 x3 = bp[5];
                    bf[t][0]  = (_Float16)x0.x; bf[t][1]  = (_Float16)x0.y;
                    bf[t][2]  = (_Float16)x0.z; bf[t][3]  = (_Float16)x0.w;
                    bf[t][4]  = (_Float16)x1.x; bf[t][5]  = (_Float16)x1.y;
                    bf[t][6]  = (_Float16)x1.z; bf[t][7]  = (_Float16)x1.w;
                    bf[t][8]  = (_Float16)x2.x; bf[t][9]  = (_Float16)x2.y;
                    bf[t][10] = (_Float16)x2.z; bf[t][11] = (_Float16)x2.w;
                    bf[t][12] = (_Float16)x3.x; bf[t][13] = (_Float16)x3.y;
                    bf[t][14] = (_Float16)x3.z; bf[t][15] = (_Float16)x3.w;
                }
#pragma unroll
                for (int t = 0; t < NT; ++t)
                    acc[t] = __builtin_amdgcn_wmma_f32_16x16x32_f16(
                        false, af, false, bf[t], (short)0, acc[t], false, false);
            }
        }
    }

    // ---- epilogue: bias + LeakyReLU(0.01); D layout VGPR j -> M=j+hi*8 ----
#pragma unroll
    for (int t = 0; t < NT; ++t) {
        const int   n  = n0 + t * 16 + l;
        const float bv = bias[n];
        float* op = out + ((size_t)b * Cout + n) * HWDIM + (size_t)(m0 + hi * 8);
#pragma unroll
        for (int j = 0; j < 8; ++j) {
            float v = acc[t][j] + bv;
            if (leaky) v = (v > 0.0f) ? v : 0.01f * v;
            op[j] = v;
        }
    }
}

// Softmax over 9 tap logits + implicit zero logit, then blend unfolded
// mem_stab patches with z. One thread per (b, c, h, w).
__global__ __launch_bounds__(256)
void fusion_softmax(const float* __restrict__ head,  // [B, 32*9, HW]
                    const float* __restrict__ mem,   // [B, 32, H, W]
                    const float* __restrict__ z,     // [B, 32, H, W]
                    float* __restrict__ out)         // [B, 32, H, W]
{
    const int idx = blockIdx.x * 256 + threadIdx.x;
    if (idx >= 8 * 32 * HWDIM) return;
    const int hw = idx & (HWDIM - 1);
    const int bc = idx >> 14;
    const int c  = bc & 31;
    const int b  = bc >> 5;
    const int h  = hw >> 7;
    const int w  = hw & (WDIM - 1);

    float lg[9];
    float mx = 0.0f;  // extra zero logit participates in the max
#pragma unroll
    for (int p = 0; p < 9; ++p) {
        lg[p] = head[((size_t)b * 288 + c * 9 + p) * HWDIM + hw];
        mx = fmaxf(mx, lg[p]);
    }
    const float ez = __expf(-mx);
    float den = ez;
    float e[9];
#pragma unroll
    for (int p = 0; p < 9; ++p) { e[p] = __expf(lg[p] - mx); den += e[p]; }

    const float* mch = mem + ((size_t)b * 32 + c) * HWDIM;
    float acc = 0.0f;
#pragma unroll
    for (int kh = 0; kh < 3; ++kh)
#pragma unroll
        for (int kw = 0; kw < 3; ++kw) {
            const int hp = h + kh - 1, wp = w + kw - 1;
            const float mv = ((unsigned)hp < (unsigned)HDIM &&
                              (unsigned)wp < (unsigned)WDIM)
                                 ? mch[hp * WDIM + wp] : 0.0f;
            acc += e[kh * 3 + kw] * mv;
        }
    acc += ez * z[((size_t)b * 32 + c) * HWDIM + hw];
    out[idx] = acc / den;
}

extern "C" void kernel_launch(void* const* d_in, const int* in_sizes, int n_in,
                              void* d_out, int out_size, void* d_ws, size_t ws_size,
                              hipStream_t stream)
{
    (void)in_sizes; (void)n_in; (void)out_size; (void)ws_size;
    const float* z        = (const float*)d_in[0];
    const float* backbone = (const float*)d_in[1];
    const float* mstab    = (const float*)d_in[2];
    const float* munstab  = (const float*)d_in[3];
    const float* w0 = (const float*)d_in[4];  const float* b0 = (const float*)d_in[5];
    const float* w1 = (const float*)d_in[6];  const float* b1 = (const float*)d_in[7];
    const float* w2 = (const float*)d_in[8];  const float* b2 = (const float*)d_in[9];
    const float* wl = (const float*)d_in[10]; const float* bl = (const float*)d_in[11];

    // workspace: q0 (8x64xHW) | q1 (8x64xHW) | head (8x288xHW); q2 aliases q0
    float* q0   = (float*)d_ws;
    float* q1   = q0 + (size_t)8 * 64 * HWDIM;
    float* head = q1 + (size_t)8 * 64 * HWDIM;

    const dim3 blk(WAVES * 32);
    const int  gx = HWDIM / (16 * WAVES);   // 256 spatial tiles per image

    // conv0: [backbone|z|mem_stab|mem_unstab] (160ch) -> 64, LeakyReLU
    conv3x3_wmma<true, 4><<<dim3(gx, 1, 8), blk, 0, stream>>>(
        backbone, z, mstab, munstab, 64, 32, 32, w0, b0, q0, 160, 64, 1);
    // conv1: 64 -> 64, LeakyReLU
    conv3x3_wmma<false, 4><<<dim3(gx, 1, 8), blk, 0, stream>>>(
        q0, q0, q0, q0, 64, 0, 0, w1, b1, q1, 64, 64, 1);
    // conv2: 64 -> 64, LeakyReLU (output overwrites q0)
    conv3x3_wmma<false, 4><<<dim3(gx, 1, 8), blk, 0, stream>>>(
        q1, q1, q1, q1, 64, 0, 0, w2, b2, q0, 64, 64, 1);
    // conv_last: 64 -> 288 (C*9), bias only; 288 = 6 * 48 -> NT=3
    conv3x3_wmma<false, 3><<<dim3(gx, 6, 8), blk, 0, stream>>>(
        q0, q0, q0, q0, 64, 0, 0, wl, bl, head, 64, 288, 0);

    const int total = 8 * 32 * HWDIM;
    fusion_softmax<<<(total + 255) / 256, 256, 0, stream>>>(
        head, mstab, z, (float*)d_out);
}
